// GatedMoE_72567767433947
// MI455X (gfx1250) — matrix operand order repaired and checked by reading, not compiled, not run on hardware
//
#include <hip/hip_runtime.h>
#include <hip/hip_bf16.h>
#include <math.h>

// ---------------- problem constants (match reference) ----------------
#define DIM  1024
#define FFN  4096
#define NE   10
#define NTOK 8192

// ---------------- tiling ----------------
constexpr int BM = 128, BN = 128, BK = 32;
constexpr int APAD = 8;
constexpr int ASTR = BK + APAD;                 // 40 bf16 row stride in LDS
constexpr int NPAD_ROWS = NTOK + NE * BM;       // 9472 (gathered rows, padded per expert)
constexpr int MT_SHARED = NTOK / BM;            // 64
constexpr int MT_EXP    = NPAD_ROWS / BM;       // 74

typedef __attribute__((ext_vector_type(16))) __bf16 v16bf;
typedef __attribute__((ext_vector_type(8)))  float  v8f;

// ---- CDNA5 async global->LDS path (guarded; sync fallback keeps compile green) ----
#if defined(__gfx1250__) && __has_builtin(__builtin_amdgcn_global_load_async_to_lds_b128)
#define HAVE_ASYNC_LDS 1
typedef int b128v __attribute__((vector_size(16)));
typedef __attribute__((address_space(1))) b128v* async_gp;   // global int4*
typedef __attribute__((address_space(3))) b128v* async_lp;   // lds int4*
#else
#define HAVE_ASYNC_LDS 0
#endif

#if defined(__gfx1250__) && __has_builtin(__builtin_amdgcn_s_wait_asynccnt)
#define ASYNC_WAIT() __builtin_amdgcn_s_wait_asynccnt(0)
#elif HAVE_ASYNC_LDS
#define ASYNC_WAIT() asm volatile("s_wait_asynccnt 0x0" ::: "memory")
#else
#define ASYNC_WAIT()
#endif

// ---------------- helpers ----------------
__device__ __forceinline__ unsigned int pack2bf(float a, float b) {
  union { __bf16 h[2]; unsigned int u; } c;
  c.h[0] = (__bf16)a; c.h[1] = (__bf16)b;
  return c.u;
}

// ---------------- small prep kernels ----------------
__global__ void k_init(int* perm, int* counts) {
  int i = blockIdx.x * 256 + threadIdx.x;
  if (i < NPAD_ROWS) perm[i] = -1;
  if (i < NE) counts[i] = 0;
}

__global__ void k_hist(const int* __restrict__ ids, int* counts) {
  int i = blockIdx.x * 256 + threadIdx.x;
  if (i < NTOK) atomicAdd(&counts[ids[i]], 1);
}

__global__ void k_scan(const int* __restrict__ counts, int* cursors, int* tile2e) {
  if (threadIdx.x != 0 || blockIdx.x != 0) return;
  int off = 0, tile = 0;
  for (int e = 0; e < NE; ++e) {
    cursors[e] = off;
    int nt = (counts[e] + BM - 1) / BM;
    for (int i = 0; i < nt && tile < MT_EXP; ++i) tile2e[tile++] = e;
    off += nt * BM;
  }
  for (; tile < MT_EXP; ++tile) tile2e[tile] = -1;
}

__global__ void k_perm(const int* __restrict__ ids, int* cursors, int* perm) {
  int i = blockIdx.x * 256 + threadIdx.x;
  if (i < NTOK) {
    int pos = atomicAdd(&cursors[ids[i]], 1);
    perm[pos] = i;
  }
}

// fp32 -> bf16 (natural token order), 4 elements/thread
__global__ void k_cvt(const float4* __restrict__ x4, __bf16* __restrict__ xb) {
  int i = blockIdx.x * 256 + threadIdx.x;
  float4 f = x4[i];
  union { __bf16 h[4]; uint2 u; } c;
  c.h[0] = (__bf16)f.x; c.h[1] = (__bf16)f.y;
  c.h[2] = (__bf16)f.z; c.h[3] = (__bf16)f.w;
  *(uint2*)&xb[(size_t)i * 4] = c.u;
}

// fp32 -> bf16 gathered by perm (pad rows -> 0), 4 elements/thread
__global__ void k_gather(const float* __restrict__ x, const int* __restrict__ perm,
                         __bf16* __restrict__ xg) {
  int i = blockIdx.x * 256 + threadIdx.x;     // group of 4 elems
  int row = i >> 8;                           // DIM/4 = 256 groups per row
  int c4  = (i & 255) * 4;
  int tok = perm[row];
  union { __bf16 h[4]; uint2 u; } c;
  if (tok >= 0) {
    float4 f = *(const float4*)&x[(size_t)tok * DIM + c4];
    c.h[0] = (__bf16)f.x; c.h[1] = (__bf16)f.y;
    c.h[2] = (__bf16)f.z; c.h[3] = (__bf16)f.w;
  } else {
    c.h[0] = c.h[1] = c.h[2] = c.h[3] = (__bf16)0.0f;
  }
  *(uint2*)&xg[(size_t)i * 4] = c.u;
}

// ---------------- fused WMMA GEMM (double-buffered, async A staging) ----------------
// EPI: 0 = +bias, GELU, store bf16 (h)
//      1 = +bias, +residual x[token], store f32 (m)
//      2 = gate: sigmoid(+bias); out[token] = g*m + (1-g)*x    (shared pass)
//      3 = gate: out[token] += g*m + (1-g)*x                   (expert pass)
template <int EPI>
__global__ void __launch_bounds__(256)
moe_gemm(const __bf16* __restrict__ A, int lda,
         const float* __restrict__ W, size_t wStride, int ldw,
         const float* __restrict__ bias, int biasStride,
         const int* __restrict__ tile2e,
         const int* __restrict__ rowmap,
         const float* __restrict__ X,
         const float* __restrict__ Mbuf,
         float* __restrict__ outF, __bf16* __restrict__ outH,
         int K, int ldo)
{
  __shared__ __align__(16) __bf16 As[2][BM * ASTR];
  __shared__ __align__(16) __bf16 Bs[2][BN * ASTR];

  const int mTile = blockIdx.x, nTile = blockIdx.y;
  int expert = 0;
  if (tile2e) { expert = tile2e[mTile]; if (expert < 0) return; }
  const float* Wt = W + (size_t)expert * wStride;
  const float* bt = bias + (size_t)expert * biasStride;

  const int t = threadIdx.x;
  const int lane  = t & 31;
  const int waveM = ((t >> 5) & 1) * 64;   // 2 waves along M
  const int waveN = (t >> 6) * 32;         // 4 waves along N

  v8f acc[4][2] = {};

  // A staging: 256 threads, 128 rows x 64B -> 2 threads/row, 2x16B each
  const int aRow = t >> 1;
  const int aOff = (t & 1) * 16;           // in bf16 elements
  // B staging: thread owns one column n, 16 consecutive k values
  const int bN = t & 127;
  const int bK = (t >> 7) * 16;

  const __bf16* Ag = A + (size_t)(mTile * BM + aRow) * lda + aOff;
  const float*  Wg = Wt + (size_t)bK * ldw + nTile * BN + bN;

  float breg[16];

  auto stage_a = [&](int k0, int buf) {
    const __bf16* src = Ag + k0;
    __bf16* dst = &As[buf][aRow * ASTR + aOff];
#if HAVE_ASYNC_LDS
    __builtin_amdgcn_global_load_async_to_lds_b128((async_gp)src, (async_lp)dst, 0, 0);
    __builtin_amdgcn_global_load_async_to_lds_b128((async_gp)src, (async_lp)dst, 16, 0);
#else
    uint4 a0 = *(const uint4*)src;
    uint4 a1 = *(const uint4*)(src + 8);
    *(uint4*)dst = a0;
    *(uint4*)(dst + 8) = a1;
#endif
  };

  auto load_b = [&](int k0) {
#pragma unroll
    for (int j = 0; j < 16; ++j) breg[j] = Wg[(size_t)(k0 + j) * ldw];
  };

  auto store_b = [&](int buf) {
    unsigned int pk[8];
#pragma unroll
    for (int j = 0; j < 8; ++j) pk[j] = pack2bf(breg[2 * j], breg[2 * j + 1]);
    *(uint4*)&Bs[buf][bN * ASTR + bK]     = *(uint4*)&pk[0];
    *(uint4*)&Bs[buf][bN * ASTR + bK + 8] = *(uint4*)&pk[4];
  };

  auto compute = [&](int buf) {
    v16bf af[4], bfv[2];
#pragma unroll
    for (int i = 0; i < 4; ++i) {
      // A 16x32 bf16 layout: lanes 0-15 m=0..15; lane>=16 picks K groups +8
      const __bf16* p = &As[buf][(waveM + i * 16 + (lane & 15)) * ASTR + (lane >> 4) * 8];
      union { uint4 q[2]; v16bf v; } u;
      u.q[0] = *(const uint4*)p;          // K = kq*8 + 0..7
      u.q[1] = *(const uint4*)(p + 16);   // K = 16 + kq*8 + 0..7
      af[i] = u.v;
    }
#pragma unroll
    for (int j = 0; j < 2; ++j) {
      // B 32x16 bf16 layout: lanes 0-15 n=0..15; lane>=16 picks K groups +8 (same as A)
      const __bf16* p = &Bs[buf][(waveN + j * 16 + (lane & 15)) * ASTR + (lane >> 4) * 8];
      union { uint4 q[2]; v16bf v; } u;
      u.q[0] = *(const uint4*)p;
      u.q[1] = *(const uint4*)(p + 16);
      bfv[j] = u.v;
    }
#pragma unroll
    for (int i = 0; i < 4; ++i)
#pragma unroll
      for (int j = 0; j < 2; ++j)
        acc[i][j] = __builtin_amdgcn_wmma_f32_16x16x32_bf16(
            false, af[i], false, bfv[j], (short)0, acc[i][j], false, false);
  };

  const int T = K / BK;

  // prologue: tile 0 into buffer 0
  stage_a(0, 0);
  load_b(0);
  store_b(0);
  ASYNC_WAIT();
  __syncthreads();

  for (int tt = 0; tt < T; ++tt) {
    const int cur = tt & 1, nxt = cur ^ 1;
    if (tt + 1 < T) {
      stage_a((tt + 1) * BK, nxt);       // async A -> LDS[nxt]
      load_b((tt + 1) * BK);             // B fp32 -> regs
      if (tt + 2 < T) {
        __builtin_prefetch(Wg + (size_t)((tt + 2) * BK) * ldw, 0, 1);
        __builtin_prefetch(Ag + (tt + 2) * BK, 0, 1);
      }
    }
    compute(cur);                        // WMMA from LDS[cur]
    if (tt + 1 < T) {
      store_b(nxt);                      // cvt fp32->bf16, ds_store to LDS[nxt]
      ASYNC_WAIT();                      // this wave's async A writes landed
    }
    __syncthreads();
  }

  // ---- epilogue ----
  // C/D layout: lane n = lane&15, vgpr r -> m = r + (lane>=16 ? 8 : 0)
  const int colB  = nTile * BN + waveN + (lane & 15);
  const int rHalf = (lane >> 4) * 8;
#pragma unroll
  for (int i = 0; i < 4; ++i) {
    const int rowB = mTile * BM + waveM + i * 16 + rHalf;
#pragma unroll
    for (int j = 0; j < 2; ++j) {
      const int col = colB + j * 16;
      const float bv = bt[col];
#pragma unroll
      for (int r = 0; r < 8; ++r) {
        const int row = rowB + r;
        float v = acc[i][j][r] + bv;
        if constexpr (EPI == 0) {
          v = 0.5f * v * (1.0f + erff(v * 0.70710678118654752f));  // exact GELU
          outH[(size_t)row * ldo + col] = (__bf16)v;
        } else if constexpr (EPI == 1) {
          const int tok = rowmap ? rowmap[row] : row;
          const float xv = (tok >= 0) ? X[(size_t)tok * DIM + col] : 0.0f;
          outF[(size_t)row * ldo + col] = v + xv;
        } else {
          const int tok = rowmap ? rowmap[row] : row;
          if (tok >= 0) {
            const float g  = 1.0f / (1.0f + __expf(-v));
            const float mv = Mbuf[(size_t)row * ldo + col];
            const float xv = X[(size_t)tok * DIM + col];
            const float val = g * mv + (1.0f - g) * xv;
            if constexpr (EPI == 3) outF[(size_t)tok * DIM + col] += val;
            else                    outF[(size_t)tok * DIM + col]  = val;
          }
        }
      }
    }
  }
}

// ---------------- launcher ----------------
extern "C" void kernel_launch(void* const* d_in, const int* in_sizes, int n_in,
                              void* d_out, int out_size, void* d_ws, size_t ws_size,
                              hipStream_t stream) {
  const float* x   = (const float*)d_in[0];
  const int*   ids = (const int*)  d_in[1];
  const float* sW1 = (const float*)d_in[2];
  const float* sb1 = (const float*)d_in[3];
  const float* sW2 = (const float*)d_in[4];
  const float* sb2 = (const float*)d_in[5];
  const float* sWg = (const float*)d_in[6];
  const float* sbg = (const float*)d_in[7];
  const float* dW1 = (const float*)d_in[8];
  const float* db1 = (const float*)d_in[9];
  const float* dW2 = (const float*)d_in[10];
  const float* db2 = (const float*)d_in[11];
  const float* dWg = (const float*)d_in[12];
  const float* dbg = (const float*)d_in[13];
  float* out = (float*)d_out;

  char* ws = (char*)d_ws;
  size_t off = 0;
  auto alloc = [&](size_t bytes) -> void* {
    void* p = ws + off;
    off += (bytes + 255) & ~(size_t)255;
    return p;
  };
  int*    perm    = (int*)   alloc((size_t)NPAD_ROWS * sizeof(int));
  int*    counts  = (int*)   alloc(NE * sizeof(int));
  int*    cursors = (int*)   alloc(NE * sizeof(int));
  int*    tile2e  = (int*)   alloc(MT_EXP * sizeof(int));
  __bf16* xb      = (__bf16*)alloc((size_t)NTOK * DIM * 2);
  __bf16* xg      = (__bf16*)alloc((size_t)NPAD_ROWS * DIM * 2);
  __bf16* hbuf    = (__bf16*)alloc((size_t)NPAD_ROWS * FFN * 2);
  float*  mbuf    = (float*) alloc((size_t)NPAD_ROWS * DIM * 4);

  // routing metadata
  k_init<<<(NPAD_ROWS + 255) / 256, 256, 0, stream>>>(perm, counts);
  k_hist<<<NTOK / 256, 256, 0, stream>>>(ids, counts);
  k_scan<<<1, 1, 0, stream>>>(counts, cursors, tile2e);
  k_perm<<<NTOK / 256, 256, 0, stream>>>(ids, cursors, perm);
  // activation conversion
  k_cvt<<<(NTOK * DIM / 4) / 256, 256, 0, stream>>>((const float4*)x, xb);
  k_gather<<<(NPAD_ROWS * DIM / 4) / 256, 256, 0, stream>>>(x, perm, xg);

  // ---- shared expert ----
  moe_gemm<0><<<dim3(MT_SHARED, FFN / BN), 256, 0, stream>>>(
      xb, DIM, sW1, 0, FFN, sb1, 0, nullptr, nullptr,
      nullptr, nullptr, nullptr, hbuf, DIM, FFN);
  moe_gemm<1><<<dim3(MT_SHARED, DIM / BN), 256, 0, stream>>>(
      hbuf, FFN, sW2, 0, DIM, sb2, 0, nullptr, nullptr,
      x, nullptr, mbuf, nullptr, FFN, DIM);
  moe_gemm<2><<<dim3(MT_SHARED, DIM / BN), 256, 0, stream>>>(
      xb, DIM, sWg, 0, DIM, sbg, 0, nullptr, nullptr,
      x, mbuf, out, nullptr, DIM, DIM);

  // ---- domain experts (gathered rows, per-tile expert weights) ----
  moe_gemm<0><<<dim3(MT_EXP, FFN / BN), 256, 0, stream>>>(
      xg, DIM, dW1, (size_t)DIM * FFN, FFN, db1, FFN, tile2e, nullptr,
      nullptr, nullptr, nullptr, hbuf, DIM, FFN);
  moe_gemm<1><<<dim3(MT_EXP, DIM / BN), 256, 0, stream>>>(
      hbuf, FFN, dW2, (size_t)FFN * DIM, DIM, db2, DIM, tile2e, perm,
      x, nullptr, mbuf, nullptr, FFN, DIM);
  moe_gemm<3><<<dim3(MT_EXP, DIM / BN), 256, 0, stream>>>(
      xg, DIM, dWg, (size_t)DIM * DIM, DIM, dbg, DIM, tile2e, perm,
      x, mbuf, out, nullptr, DIM, DIM);
}
